// LiteSelfAttention2D_20916490731872
// MI455X (gfx1250) — compile-verified, matrix operand order checked by reading, not run on hardware
//
#include <hip/hip_runtime.h>
#include <hip/hip_bf16.h>

// ---------------------------------------------------------------------------
// LiteSelfAttention2D for MI455X (gfx1250): flash-attention on v_wmma_f32_16x16x32_f16
// B=4, C=256, KEY_CH=128, HEADS=4, HEAD_DIM=32, N=4096
// ---------------------------------------------------------------------------

#define NB      4
#define IN_CH   256
#define KEY_CH  128
#define HEADS   4
#define HEAD_DIM 32
#define NPIX    4096   // 64*64

typedef _Float16 half8  __attribute__((ext_vector_type(8)));
typedef _Float16 half16 __attribute__((ext_vector_type(16)));
typedef float    f32x8  __attribute__((ext_vector_type(8)));

// Build a 16x32 (or 32x16) f16 WMMA fragment from two 16-byte chunks.
// A-layout: lane L holds row L%16, K-chunks at g*8 and 16+g*8 (g = L/16).
// B-layout is the mirror (lane = column, same K pattern).
static __device__ __forceinline__ half16 load_frag(const _Float16* lo,
                                                   const _Float16* hi) {
  half8 a = *(const half8*)lo;
  half8 b = *(const half8*)hi;
  half16 r;
#pragma unroll
  for (int i = 0; i < 8; ++i) { r[i] = a[i]; r[i + 8] = b[i]; }
  return r;
}

static __device__ __forceinline__ f32x8 wmma_f16(half16 a, half16 b, f32x8 c) {
  // D = A(16x32 f16) * B(32x16 f16) + C(16x16 f32)
  return __builtin_amdgcn_wmma_f32_16x16x32_f16(false, a, false, b,
                                                (short)0, c, false, false);
}

static __device__ __forceinline__ f32x8 zero8() {
  f32x8 z = {0.f, 0.f, 0.f, 0.f, 0.f, 0.f, 0.f, 0.f};
  return z;
}

// ---------------------------------------------------------------------------
// Kernel 0a: x (f32, [B][C][N]) -> xh (f16), vectorized
// ---------------------------------------------------------------------------
__global__ void __launch_bounds__(256) cvt_x_kernel(const float* __restrict__ x,
                                                    _Float16* __restrict__ xh) {
  int i = (blockIdx.x * 256 + threadIdx.x) * 4;
  float4 v = *(const float4*)(x + i);
  xh[i + 0] = (_Float16)v.x;
  xh[i + 1] = (_Float16)v.y;
  xh[i + 2] = (_Float16)v.z;
  xh[i + 3] = (_Float16)v.w;
}

// ---------------------------------------------------------------------------
// Kernel 0b: weights -> f16. Softmax scale 1/sqrt(32) is folded into Wq.
// ---------------------------------------------------------------------------
__global__ void __launch_bounds__(256) cvt_w_kernel(
    const float* __restrict__ wq, const float* __restrict__ wk,
    const float* __restrict__ wv, const float* __restrict__ wp,
    _Float16* __restrict__ wqh, _Float16* __restrict__ wkh,
    _Float16* __restrict__ wvh, _Float16* __restrict__ wph) {
  int i = blockIdx.x * 256 + threadIdx.x;
  const float s = 0.17677669529663687f;  // 1/sqrt(HEAD_DIM)
  wqh[i] = (_Float16)(wq[i] * s);
  wkh[i] = (_Float16)wk[i];
  wvh[i] = (_Float16)wv[i];
  wph[i] = (_Float16)wp[i];
}

// ---------------------------------------------------------------------------
// Kernel 1: projections Q,K,V = W * x  (per image: [128,256] x [256,4096])
// Grid: NB*64 blocks, 256 threads (8 waves). Wave w -> output rows 16w..16w+15.
// x tiles are transposed through LDS so the B-fragment (K = channel dim)
// becomes two contiguous ds_load_b128 per lane.
// Outputs: Qh,Kh f16 [b][h][n][32] (row-major), Vt f16 [b][h][32][n].
// ---------------------------------------------------------------------------
__global__ void __launch_bounds__(256) proj_qkv_kernel(
    const _Float16* __restrict__ xh,
    const _Float16* __restrict__ wqh, const _Float16* __restrict__ wkh,
    const _Float16* __restrict__ wvh,
    _Float16* __restrict__ Qh, _Float16* __restrict__ Kh,
    _Float16* __restrict__ Vt) {
  __shared__ __align__(16) _Float16 Xt[64][40];  // [n][c], padded stride 40

  const int bx = blockIdx.x;
  const int b = bx >> 6;
  const int n0 = (bx & 63) * 64;
  const int tid = threadIdx.x;
  const int w = tid >> 5, lane = tid & 31, g = lane >> 4, lh = lane & 15;
  const _Float16* xb = xh + (size_t)b * IN_CH * NPIX;

  f32x8 acc[3][4];
#pragma unroll
  for (int p = 0; p < 3; ++p)
#pragma unroll
    for (int j = 0; j < 4; ++j) acc[p][j] = zero8();

  const int wrow = w * 16 + lh;  // output row for this lane's fragment
  for (int cc = 0; cc < IN_CH; cc += 32) {
    __syncthreads();
    // load [32 c][64 n] tile, store transposed as [n][c]
#pragma unroll
    for (int k = 0; k < 8; ++k) {
      int idx = k * 256 + tid;          // 0..2047
      int c = idx >> 6, n = idx & 63;
      Xt[n][c] = xb[(size_t)(cc + c) * NPIX + n0 + n];
    }
    __syncthreads();

    half16 aq = load_frag(wqh + wrow * IN_CH + cc + g * 8,
                          wqh + wrow * IN_CH + cc + 16 + g * 8);
    half16 ak = load_frag(wkh + wrow * IN_CH + cc + g * 8,
                          wkh + wrow * IN_CH + cc + 16 + g * 8);
    half16 av = load_frag(wvh + wrow * IN_CH + cc + g * 8,
                          wvh + wrow * IN_CH + cc + 16 + g * 8);
#pragma unroll
    for (int j = 0; j < 4; ++j) {
      half16 bf = load_frag(&Xt[j * 16 + lh][g * 8],
                            &Xt[j * 16 + lh][16 + g * 8]);
      acc[0][j] = wmma_f16(aq, bf, acc[0][j]);
      acc[1][j] = wmma_f16(ak, bf, acc[1][j]);
      acc[2][j] = wmma_f16(av, bf, acc[2][j]);
    }
  }

  // scatter to Qh/Kh (row-major per pixel) and Vt (transposed)
#pragma unroll
  for (int j = 0; j < 4; ++j) {
#pragma unroll
    for (int r = 0; r < 8; ++r) {
      int kk = w * 16 + r + 8 * g;       // 0..127
      int h = kk >> 5, d = kk & 31;
      int n = n0 + j * 16 + lh;
      size_t hb = (size_t)(b * HEADS + h);
      size_t qoff = (hb * NPIX + n) * HEAD_DIM + d;
      Qh[qoff] = (_Float16)acc[0][j][r];
      Kh[qoff] = (_Float16)acc[1][j][r];
      Vt[(hb * HEAD_DIM + d) * NPIX + n] = (_Float16)acc[2][j][r];
    }
  }
}

// ---------------------------------------------------------------------------
// Kernel 2: flash attention. Grid (N/128, B*HEADS), 256 threads = 8 waves.
// Each wave owns 16 query rows. Per 64-key tile:
//   - all 8 waves cooperatively stage the SHARED K tile (64x32 f16, 4 KB) and
//     V^T tile (32x64 f16, 4 KB) into LDS once (8x less global/L2 traffic
//     than per-wave streaming), with global_prefetch of the next tile;
//   - 4 WMMAs for S = Q K^T (scale pre-folded into Q),
//   - online softmax in f32 through per-wave LDS (lanes 0..15 own rows),
//   - 4 WMMAs for O += P V (P re-read from LDS in A-layout).
// Output Orow f16 [b][n][128] (channel-contiguous per pixel).
// ---------------------------------------------------------------------------
__global__ void __launch_bounds__(256) attn_kernel(
    const _Float16* __restrict__ Qh, const _Float16* __restrict__ Kh,
    const _Float16* __restrict__ Vt, _Float16* __restrict__ Orow) {
  __shared__ __align__(16) float    Ss[8][16][64];   // 32 KB (per-wave S)
  __shared__ __align__(16) _Float16 Ps[8][16][64];   // 16 KB (per-wave P)
  __shared__ __align__(16) _Float16 Ks[64][32];      //  4 KB (shared K tile)
  __shared__ __align__(16) _Float16 Vs[32][64];      //  4 KB (shared V^T tile)
  __shared__ float Mrow[8][16], Lrow[8][16], Arow[8][16];

  const int tid = threadIdx.x;
  const int w = tid >> 5, lane = tid & 31, g = lane >> 4, lh = lane & 15;
  const int bh = blockIdx.y;
  const int b = bh >> 2, h = bh & 3;
  const int q0 = blockIdx.x * 128 + w * 16;
  const size_t hb = (size_t)(b * HEADS + h);
  const _Float16* qb = Qh + hb * NPIX * HEAD_DIM;
  const _Float16* kb = Kh + hb * NPIX * HEAD_DIM;
  const _Float16* vb = Vt + hb * HEAD_DIM * NPIX;

  // Q fragment for this wave's 16 rows (reused every iteration)
  half16 aq = load_frag(qb + (size_t)(q0 + lh) * 32 + g * 8,
                        qb + (size_t)(q0 + lh) * 32 + 16 + g * 8);
  f32x8 o_lo = zero8(), o_hi = zero8();
  if (lane < 16) { Mrow[w][lane] = -1e30f; Lrow[w][lane] = 0.f; }

  // cooperative-staging indices (each thread moves 8+8 f16 per tile)
  const int kslot = tid * 8;            // 0..2047 over Ks (64x32)
  const int vd = tid >> 3;              // 0..31  (V^T row = d)
  const int vm = (tid & 7) * 8;         // 0..56  (m offset within tile)

#pragma unroll 1
  for (int kt = 0; kt < NPIX / 64; ++kt) {
    const int m0 = kt * 64;
    // ---- stage K/V tiles into LDS once per workgroup ----
    __syncthreads();  // previous iteration's readers are done
    *(half8*)((_Float16*)Ks + kslot) = *(const half8*)(kb + (size_t)m0 * 32 + kslot);
    *(half8*)(&Vs[vd][vm])           = *(const half8*)(vb + (size_t)vd * NPIX + m0 + vm);
    if (kt + 1 < NPIX / 64) {           // prefetch next tile (global_prefetch_b8)
      __builtin_prefetch(kb + (size_t)(m0 + 64) * 32 + kslot, 0, 0);
      __builtin_prefetch(vb + (size_t)vd * NPIX + m0 + 64 + vm, 0, 0);
    }
    __syncthreads();  // staging visible to all waves

    // ---- S = Q K^T over 64 keys (B-fragments from shared LDS tile) ----
#pragma unroll
    for (int j = 0; j < 4; ++j) {
      half16 bk = load_frag(&Ks[j * 16 + lh][g * 8],
                            &Ks[j * 16 + lh][16 + g * 8]);
      f32x8 s = wmma_f16(aq, bk, zero8());
#pragma unroll
      for (int r = 0; r < 8; ++r) Ss[w][r + 8 * g][j * 16 + lh] = s[r];
    }
    // ---- online softmax: lanes 0..15 each own one query row ----
    if (lane < 16) {
      const float4* row = (const float4*)&Ss[w][lane][0];
      float mold = Mrow[w][lane];
      float mx = mold;
#pragma unroll
      for (int jj = 0; jj < 16; ++jj) {
        float4 v = row[jj];
        mx = fmaxf(mx, fmaxf(fmaxf(v.x, v.y), fmaxf(v.z, v.w)));
      }
      float alpha = __expf(mold - mx);
      float ssum = 0.f;
#pragma unroll
      for (int jj = 0; jj < 16; ++jj) {
        float4 v = row[jj];
        float p0 = __expf(v.x - mx), p1 = __expf(v.y - mx);
        float p2 = __expf(v.z - mx), p3 = __expf(v.w - mx);
        ssum += (p0 + p1) + (p2 + p3);
        Ps[w][lane][jj * 4 + 0] = (_Float16)p0;
        Ps[w][lane][jj * 4 + 1] = (_Float16)p1;
        Ps[w][lane][jj * 4 + 2] = (_Float16)p2;
        Ps[w][lane][jj * 4 + 3] = (_Float16)p3;
      }
      Mrow[w][lane] = mx;
      Lrow[w][lane] = Lrow[w][lane] * alpha + ssum;
      Arow[w][lane] = alpha;
    }
    // ---- rescale running O by alpha(row) ----
#pragma unroll
    for (int r = 0; r < 8; ++r) {
      float a = Arow[w][r + 8 * g];
      o_lo[r] *= a;
      o_hi[r] *= a;
    }
    // ---- O += P V (two 32-wide key sub-tiles; d split 0..15 / 16..31) ----
#pragma unroll
    for (int mj = 0; mj < 2; ++mj) {
      half16 ap = load_frag(&Ps[w][lh][mj * 32 + g * 8],
                            &Ps[w][lh][mj * 32 + 16 + g * 8]);
      half16 bv0 = load_frag(&Vs[lh][mj * 32 + g * 8],
                             &Vs[lh][mj * 32 + 16 + g * 8]);
      o_lo = wmma_f16(ap, bv0, o_lo);
      half16 bv1 = load_frag(&Vs[16 + lh][mj * 32 + g * 8],
                             &Vs[16 + lh][mj * 32 + 16 + g * 8]);
      o_hi = wmma_f16(ap, bv1, o_hi);
    }
  }

  // ---- finalize: divide by l, store Orow[b][n][h*32 + d] ----
  _Float16* ob = Orow + (size_t)b * NPIX * KEY_CH + h * HEAD_DIM;
#pragma unroll
  for (int r = 0; r < 8; ++r) {
    int n = q0 + r + 8 * g;
    float inv = 1.f / Lrow[w][r + 8 * g];
    ob[(size_t)n * KEY_CH + lh]      = (_Float16)(o_lo[r] * inv);
    ob[(size_t)n * KEY_CH + 16 + lh] = (_Float16)(o_hi[r] * inv);
  }
}

// ---------------------------------------------------------------------------
// Kernel 3: out = Wp @ O + x.  [256,128] x [128,4096] per image.
// Orow rows are kk-contiguous, so B-fragments are direct global_load_b128.
// Grid (N/64, B, 2), 256 threads; wave w handles 16 output channels.
// ---------------------------------------------------------------------------
__global__ void __launch_bounds__(256) out_proj_kernel(
    const _Float16* __restrict__ wph, const _Float16* __restrict__ Orow,
    const float* __restrict__ x, float* __restrict__ out) {
  const int tid = threadIdx.x;
  const int w = tid >> 5, lane = tid & 31, g = lane >> 4, lh = lane & 15;
  const int n0 = blockIdx.x * 64;
  const int b = blockIdx.y;
  const int c0 = blockIdx.z * 128 + w * 16;
  const _Float16* ob = Orow + (size_t)b * NPIX * KEY_CH;

  f32x8 acc[4];
#pragma unroll
  for (int j = 0; j < 4; ++j) acc[j] = zero8();

#pragma unroll
  for (int kc = 0; kc < 4; ++kc) {
    const int k0 = kc * 32;
    half16 a = load_frag(wph + (size_t)(c0 + lh) * KEY_CH + k0 + g * 8,
                         wph + (size_t)(c0 + lh) * KEY_CH + k0 + 16 + g * 8);
#pragma unroll
    for (int j = 0; j < 4; ++j) {
      const _Float16* orow = ob + (size_t)(n0 + j * 16 + lh) * KEY_CH + k0;
      half16 bf = load_frag(orow + g * 8, orow + 16 + g * 8);
      acc[j] = wmma_f16(a, bf, acc[j]);
    }
  }
#pragma unroll
  for (int j = 0; j < 4; ++j) {
#pragma unroll
    for (int r = 0; r < 8; ++r) {
      int c = c0 + r + 8 * g;
      int n = n0 + j * 16 + lh;
      size_t off = (size_t)(b * IN_CH + c) * NPIX + n;
      out[off] = acc[j][r] + x[off];
    }
  }
}

// ---------------------------------------------------------------------------
// Host launcher
// ---------------------------------------------------------------------------
extern "C" void kernel_launch(void* const* d_in, const int* in_sizes, int n_in,
                              void* d_out, int out_size, void* d_ws, size_t ws_size,
                              hipStream_t stream) {
  const float* x  = (const float*)d_in[0];
  const float* Wq = (const float*)d_in[1];
  const float* Wk = (const float*)d_in[2];
  const float* Wv = (const float*)d_in[3];
  const float* Wp = (const float*)d_in[4];
  float* out = (float*)d_out;

  // workspace carve-up (all f16)
  char* ws = (char*)d_ws;
  size_t off = 0;
  _Float16* xh  = (_Float16*)(ws + off); off += (size_t)NB * IN_CH * NPIX * 2;   // 8 MB
  _Float16* wqh = (_Float16*)(ws + off); off += (size_t)KEY_CH * IN_CH * 2;
  _Float16* wkh = (_Float16*)(ws + off); off += (size_t)KEY_CH * IN_CH * 2;
  _Float16* wvh = (_Float16*)(ws + off); off += (size_t)KEY_CH * IN_CH * 2;
  _Float16* wph = (_Float16*)(ws + off); off += (size_t)IN_CH * KEY_CH * 2;
  _Float16* Qh  = (_Float16*)(ws + off); off += (size_t)NB * KEY_CH * NPIX * 2;  // 4 MB
  _Float16* Kh  = (_Float16*)(ws + off); off += (size_t)NB * KEY_CH * NPIX * 2;  // 4 MB
  _Float16* Vt  = (_Float16*)(ws + off); off += (size_t)NB * KEY_CH * NPIX * 2;  // 4 MB
  _Float16* Orow = (_Float16*)(ws + off); off += (size_t)NB * NPIX * KEY_CH * 2; // 4 MB

  // 0) precision conversion (scale folded into Wq)
  cvt_x_kernel<<<(NB * IN_CH * NPIX) / (256 * 4), 256, 0, stream>>>(x, xh);
  cvt_w_kernel<<<(KEY_CH * IN_CH) / 256, 256, 0, stream>>>(
      Wq, Wk, Wv, Wp, wqh, wkh, wvh, wph);

  // 1) Q/K/V projections
  proj_qkv_kernel<<<NB * (NPIX / 64), 256, 0, stream>>>(
      xh, wqh, wkh, wvh, Qh, Kh, Vt);

  // 2) flash attention
  attn_kernel<<<dim3(NPIX / 128, NB * HEADS), 256, 0, stream>>>(Qh, Kh, Vt, Orow);

  // 3) output projection + residual
  out_proj_kernel<<<dim3(NPIX / 64, NB, 2), 256, 0, stream>>>(wph, Orow, x, out);
}
